// CorrelationGuidedAttention_27900107555119
// MI455X (gfx1250) — matrix operand order repaired and checked by reading, not compiled, not run on hardware
//
#include <hip/hip_runtime.h>

typedef __attribute__((ext_vector_type(16))) _Float16 v16h;
typedef __attribute__((ext_vector_type(8)))  _Float16 v8h;
typedef __attribute__((ext_vector_type(8)))  float    v8f;
typedef float f4 __attribute__((ext_vector_type(4), aligned(4)));

#define NF    19
#define NN    361
#define PAD   32
#define WAVES 8
#define LOG2E 1.44269504088896340736f

// 19-elem dot(row, fj)+acc; global b128 chunks, dword-aligned, tail overlapped at +15
__device__ __forceinline__ float dot19(const float* __restrict__ row,
                                       const float* fj, float acc) {
    f4 c0 = *(const f4*)(row + 0);
    f4 c1 = *(const f4*)(row + 4);
    f4 c2 = *(const f4*)(row + 8);
    f4 c3 = *(const f4*)(row + 12);
    f4 c4 = *(const f4*)(row + 15);   // .y/.z/.w = elements 16,17,18
    acc += fj[0]*c0.x;  acc += fj[1]*c0.y;  acc += fj[2]*c0.z;  acc += fj[3]*c0.w;
    acc += fj[4]*c1.x;  acc += fj[5]*c1.y;  acc += fj[6]*c1.z;  acc += fj[7]*c1.w;
    acc += fj[8]*c2.x;  acc += fj[9]*c2.y;  acc += fj[10]*c2.z; acc += fj[11]*c2.w;
    acc += fj[12]*c3.x; acc += fj[13]*c3.y; acc += fj[14]*c3.z; acc += fj[15]*c3.w;
    acc += fj[16]*c4.y; acc += fj[17]*c4.z; acc += fj[18]*c4.w;
    return acc;
}

__device__ __forceinline__ float sum19(const float* __restrict__ row) {
    f4 c0 = *(const f4*)(row + 0);
    f4 c1 = *(const f4*)(row + 4);
    f4 c2 = *(const f4*)(row + 8);
    f4 c3 = *(const f4*)(row + 12);
    f4 c4 = *(const f4*)(row + 15);
    return c0.x+c0.y+c0.z+c0.w + c1.x+c1.y+c1.z+c1.w + c2.x+c2.y+c2.z+c2.w
         + c3.x+c3.y+c3.z+c3.w + c4.y+c4.z+c4.w;
}

__global__ __launch_bounds__(256)
void cga_wmma_kernel(const float* __restrict__ features,
                     const float* __restrict__ Wq, const float* __restrict__ bq,
                     const float* __restrict__ Wk, const float* __restrict__ bk,
                     const float* __restrict__ Wv, const float* __restrict__ bv,
                     const float* __restrict__ Wo, const float* __restrict__ bo,
                     const float* __restrict__ corr,
                     const float* __restrict__ temperature,
                     float* __restrict__ out, int Btot)
{
    __shared__ _Float16 WoH[PAD * PAD];          // Wo row-major, zero-padded (B fragments)
    __shared__ _Float16 AH[WAVES][PAD * PAD];    // per-wave normalized attended rows, f16
    __shared__ float    KV[WAVES][40];           // k in [0..18], v in [20..38] (16B-chunked)
    __shared__ float    DC[WAVES][384];          // per-wave compacted 19x19 output tile

    const int tid  = threadIdx.x;
    const int lane = tid & 31;
    const int wave = tid >> 5;
    const int b    = blockIdx.x * WAVES + wave;
    const bool bok = (b < Btot);

    for (int idx = tid; idx < PAD * PAD; idx += 256) {
        int o = idx >> 5, j = idx & 31;
        WoH[idx] = (o < NF && j < NF) ? (_Float16)Wo[o * NF + j] : (_Float16)0.0f;
    }

    // ---- features for this (wave-uniform) batch via scalar loads -> SGPR array ----
    const int bu = __builtin_amdgcn_readfirstlane(bok ? b : 0);
    const float* fp = features + (size_t)bu * NF;
    f4 u0 = *(const f4*)(fp + 0);
    f4 u1 = *(const f4*)(fp + 4);
    f4 u2 = *(const f4*)(fp + 8);
    f4 u3 = *(const f4*)(fp + 12);
    f4 u4 = *(const f4*)(fp + 15);
    const float fj[NF] = { u0.x,u0.y,u0.z,u0.w, u1.x,u1.y,u1.z,u1.w,
                           u2.x,u2.y,u2.z,u2.w, u3.x,u3.y,u3.z,u3.w,
                           u4.y,u4.z,u4.w };

    // ---- lane i owns row i (rows >=19 compute harmless garbage, never escapes) ----
    const int i  = lane;
    const int ri = (i < NF) ? i : 0;                       // clamp to stay in-bounds
    float q = dot19(Wq + ri * NF, fj, bq[ri]);
    float k = dot19(Wk + ri * NF, fj, bk[ri]);
    float v = dot19(Wv + ri * NF, fj, bv[ri]);
    float colsum = sum19(corr + ri * NF);                  // corr symmetric: colsum == rowsum

    // cw_mean = (1/N) * sum_i K[i]*colsum[i]   (wave32 reduction)
    float part = (i < NF) ? k * colsum : 0.0f;
    #pragma unroll
    for (int off = 16; off; off >>= 1) part += __shfl_xor(part, off);
    const float cw  = part * (1.0f / (float)NF);
    // fold 1/T and log2(e) into the per-row scale; softmax done in exp2 domain
    const float sI2 = q * (cw * (LOG2E / temperature[0]));

    // broadcast k,v to all lanes through LDS (aligned 16B chunks, slot 19/39 dead)
    if (i < NF) { KV[wave][i] = k; KV[wave][20 + i] = v; }
    __syncthreads();
    const float* kvp = KV[wave];
    f4 a0c = *(const f4*)(kvp + 0),  a1c = *(const f4*)(kvp + 4);
    f4 a2c = *(const f4*)(kvp + 8),  a3c = *(const f4*)(kvp + 12);
    f4 a4c = *(const f4*)(kvp + 16);                       // .x/.y/.z = k16..k18
    f4 b0c = *(const f4*)(kvp + 20), b1c = *(const f4*)(kvp + 24);
    f4 b2c = *(const f4*)(kvp + 28), b3c = *(const f4*)(kvp + 32);
    f4 b4c = *(const f4*)(kvp + 36);                       // .x/.y/.z = v16..v18
    const float kx[NF] = { a0c.x,a0c.y,a0c.z,a0c.w, a1c.x,a1c.y,a1c.z,a1c.w,
                           a2c.x,a2c.y,a2c.z,a2c.w, a3c.x,a3c.y,a3c.z,a3c.w,
                           a4c.x,a4c.y,a4c.z };
    const float vx[NF] = { b0c.x,b0c.y,b0c.z,b0c.w, b1c.x,b1c.y,b1c.z,b1c.w,
                           b2c.x,b2c.y,b2c.z,b2c.w, b3c.x,b3c.y,b3c.z,b3c.w,
                           b4c.x,b4c.y,b4c.z };

    // ---- softmax row i: exp2-domain, 1/Z folded in before f16 pack ----
    float tj[NF];
    float m = -3.4e38f;
    #pragma unroll
    for (int j = 0; j < NF; ++j) { tj[j] = sI2 * kx[j]; m = fmaxf(m, tj[j]); }
    float Z = 0.0f;
    #pragma unroll
    for (int j = 0; j < NF; ++j) {
        float e = __builtin_exp2f(tj[j] - m);
        Z += e;
        tj[j] = e * vx[j];
    }
    const float iZ = 1.0f / Z;

    v16h r0, r1;
    #pragma unroll
    for (int p = 0; p < 8; ++p) {                          // v_cvt_pk_rtz_f16_f32 pairs
        auto pk = __builtin_amdgcn_cvt_pkrtz(tj[2*p] * iZ, tj[2*p+1] * iZ);
        r0[2*p] = (_Float16)pk[0]; r0[2*p+1] = (_Float16)pk[1];
    }
    #pragma unroll
    for (int p = 0; p < 16; ++p) r1[p] = (_Float16)0.0f;   // K-pad cols 19..31 = 0
    {
        auto pk = __builtin_amdgcn_cvt_pkrtz(tj[16] * iZ, tj[17] * iZ);
        r1[0] = (_Float16)pk[0]; r1[1] = (_Float16)pk[1];
        r1[2] = (_Float16)(tj[18] * iZ);
    }
    _Float16* AHw = AH[wave];
    *(v16h*)&AHw[i * PAD]      = r0;                       // two 32-B LDS stores
    *(v16h*)&AHw[i * PAD + 16] = r1;

    // bias + residual per output column o (lane i holds bo[i]+feat[i], gathered by shfl)
    const float fres = (i < NF && bok) ? features[(size_t)b * NF + i] : 0.0f;
    const float bfr  = ((i < NF) ? bo[ri] : 0.0f) + fres;
    __syncthreads();

    // ---- 32x32 GEMM: four v_wmma_f32_16x16x32_f16; bias+residual ride in C ----
    const int hgrp = lane >> 4;
    const int nloc = lane & 15;
    const int akb  = hgrp ? 16 : 0;                        // 16-bit A fragment K base
    const int bkb  = hgrp ? 16 : 0;                        // 16-bit B fragment K base

    v16h fa0, fa1;
    {
        const v8h l0 = *(const v8h*)&AHw[nloc * PAD + akb];
        const v8h h0 = *(const v8h*)&AHw[nloc * PAD + akb + 8];
        const v8h l1 = *(const v8h*)&AHw[(16 + nloc) * PAD + akb];
        const v8h h1 = *(const v8h*)&AHw[(16 + nloc) * PAD + akb + 8];
        #pragma unroll
        for (int t = 0; t < 8; ++t) {
            fa0[t] = l0[t]; fa0[8 + t] = h0[t];
            fa1[t] = l1[t]; fa1[8 + t] = h1[t];
        }
    }
    const v16h fb0 = *(const v16h*)&WoH[nloc * PAD + bkb];
    const v16h fb1 = *(const v16h*)&WoH[(16 + nloc) * PAD + bkb];

    const float add0 = __shfl(bfr, nloc);
    const float add1 = __shfl(bfr, 16 + nloc);
    v8f cb0, cb1;
    #pragma unroll
    for (int r = 0; r < 8; ++r) { cb0[r] = add0; cb1[r] = add1; }

    v8f c00 = __builtin_amdgcn_wmma_f32_16x16x32_f16(false, fa0, false, fb0, (short)0, cb0, false, false);
    v8f c01 = __builtin_amdgcn_wmma_f32_16x16x32_f16(false, fa0, false, fb1, (short)0, cb1, false, false);
    v8f c10 = __builtin_amdgcn_wmma_f32_16x16x32_f16(false, fa1, false, fb0, (short)0, cb0, false, false);
    v8f c11 = __builtin_amdgcn_wmma_f32_16x16x32_f16(false, fa1, false, fb1, (short)0, cb1, false, false);

    // ---- compacted D staging: only the valid 19x19 region written/read ----
    float* DCw = DC[wave];
    {   // tile (0,0): rows r + 8*hgrp, col nloc
        float* p = DCw + (hgrp * 8) * NF + nloc;
        #pragma unroll
        for (int r = 0; r < 8; ++r) p[r * NF] = c00[r];
    }
    if (nloc < 3) {   // tile (0,1): col 16+nloc
        float* p = DCw + (hgrp * 8) * NF + 16 + nloc;
        #pragma unroll
        for (int r = 0; r < 8; ++r) p[r * NF] = c01[r];
    }
    if (hgrp == 0) {  // tile (1,0): rows 16..18
        float* p = DCw + 16 * NF + nloc;
        #pragma unroll
        for (int r = 0; r < 3; ++r) p[r * NF] = c10[r];
    }
    if (hgrp == 0 && nloc < 3) {  // tile (1,1): rows 16..18, cols 16..18
        float* p = DCw + 16 * NF + 16 + nloc;
        #pragma unroll
        for (int r = 0; r < 3; ++r) p[r * NF] = c11[r];
    }
    __syncthreads();

    // ---- coalesced copy: 90 x b128 + 1 dword = 361 floats ----
    if (bok) {
        float* og = out + (size_t)b * NN;
        const f4* DCv = (const f4*)DCw;
        *(f4*)(og + lane * 4)        = DCv[lane];
        *(f4*)(og + (lane + 32) * 4) = DCv[lane + 32];
        if (lane < 26) *(f4*)(og + (lane + 64) * 4) = DCv[lane + 64];
        if (lane == 0) og[360] = DCw[360];
    }
}

extern "C" void kernel_launch(void* const* d_in, const int* in_sizes, int n_in,
                              void* d_out, int out_size, void* d_ws, size_t ws_size,
                              hipStream_t stream) {
    const float* features    = (const float*)d_in[0];
    const float* Wq          = (const float*)d_in[1];
    const float* bq          = (const float*)d_in[2];
    const float* Wk          = (const float*)d_in[3];
    const float* bk          = (const float*)d_in[4];
    const float* Wv          = (const float*)d_in[5];
    const float* bv          = (const float*)d_in[6];
    const float* Wo          = (const float*)d_in[7];
    const float* bo          = (const float*)d_in[8];
    const float* corr        = (const float*)d_in[9];
    const float* temperature = (const float*)d_in[10];

    const int B = in_sizes[0] / NF;
    const int blocks = (B + WAVES - 1) / WAVES;
    cga_wmma_kernel<<<blocks, 256, 0, stream>>>(features, Wq, bq, Wk, bk, Wv, bv,
                                                Wo, bo, corr, temperature,
                                                (float*)d_out, B);
}
